// GAT_37263136260426
// MI455X (gfx1250) — compile-verified, matrix operand order checked by reading, not run on hardware
//
#include <hip/hip_runtime.h>
#include <math.h>

#define NNODES 100000
#define NEDGES 1600000
#define NEG_SLOPE 0.2f
#define KDIM 128

typedef __attribute__((ext_vector_type(16))) __bf16        v16bf;
typedef __attribute__((ext_vector_type(8)))  float         v8f;
typedef __attribute__((ext_vector_type(4)))  unsigned int  u32x4;

__device__ inline unsigned short f2bf(float f) {
  unsigned int u = __float_as_uint(f);
  u += 0x7FFFu + ((u >> 16) & 1u);     // round-to-nearest-even
  return (unsigned short)(u >> 16);
}

__device__ inline void atomicMaxF(float* addr, float v) {
  if (v >= 0.f) atomicMax(reinterpret_cast<int*>(addr), __float_as_int(v));
  else          atomicMin(reinterpret_cast<unsigned int*>(addr), __float_as_uint(v));
}

// ---------------- utility kernels ----------------
__global__ void fill_f32(float* __restrict__ p, float v, long n) {
  long i = (long)blockIdx.x * blockDim.x + threadIdx.x;
  if (i < n) p[i] = v;
}

__global__ void cvt_bf16(const float* __restrict__ in, unsigned short* __restrict__ out, long n) {
  long i = (long)blockIdx.x * blockDim.x + threadIdx.x;
  if (i < n) out[i] = f2bf(in[i]);
}

// Wt[o*in_dim + k] = bf16(W[k*out_dim + o])
__global__ void transpose_cvt(const float* __restrict__ W, unsigned short* __restrict__ Wt,
                              int in_dim, int out_dim) {
  int i = blockIdx.x * blockDim.x + threadIdx.x;
  if (i >= in_dim * out_dim) return;
  int o = i / in_dim, k = i - o * in_dim;
  Wt[(size_t)o * in_dim + k] = f2bf(W[(size_t)k * out_dim + o]);
}

// ---------------- WMMA GEMM: C[M,Nout] = A[M,128] * Bt[Nout,128]^T ----------------
template<int NT>
__global__ __launch_bounds__(128)
void gemm_bf16(const unsigned short* __restrict__ A, const unsigned short* __restrict__ Bt,
               float* __restrict__ C, int M, int Nout) {
  const int lane = threadIdx.x & 31;
  const int wave = threadIdx.x >> 5;
  const int row0 = (blockIdx.x * 4 + wave) * 16;
  if (row0 >= M) return;
  const int col0 = blockIdx.y * (NT * 16);
  const int mr = lane & 15;
  const int kb = (lane >> 4) * 8;       // K-base: lanes 0-15 -> 0, lanes 16-31 -> 8

  v8f acc[NT];
#pragma unroll
  for (int t = 0; t < NT; ++t)
#pragma unroll
    for (int j = 0; j < 8; ++j) acc[t][j] = 0.f;

  const unsigned short* arow = A + (size_t)(row0 + mr) * KDIM;
#pragma unroll
  for (int kc = 0; kc < KDIM; kc += 32) {
    union { v16bf v; u32x4 q[2]; } fa;
    fa.q[0] = *reinterpret_cast<const u32x4*>(arow + kc + kb);        // K = kb..kb+7
    fa.q[1] = *reinterpret_cast<const u32x4*>(arow + kc + kb + 16);   // K = kb+16..kb+23
#pragma unroll
    for (int t = 0; t < NT; ++t) {
      const unsigned short* brow = Bt + (size_t)(col0 + t * 16 + mr) * KDIM;
      union { v16bf v; u32x4 q[2]; } fb;
      fb.q[0] = *reinterpret_cast<const u32x4*>(brow + kc + kb);
      fb.q[1] = *reinterpret_cast<const u32x4*>(brow + kc + kb + 16);
      acc[t] = __builtin_amdgcn_wmma_f32_16x16x32_bf16(false, fa.v, false, fb.v,
                                                       (short)0, acc[t], false, false);
    }
  }
  const int mbase = row0 + ((lane >> 4) ? 8 : 0);
#pragma unroll
  for (int t = 0; t < NT; ++t)
#pragma unroll
    for (int j = 0; j < 8; ++j)
      C[(size_t)(mbase + j) * Nout + col0 + t * 16 + mr] = acc[t][j];
}

// ---------------- attention logits per node: el/er[n,h] = sum_d feat*a ----------------
__global__ void node_attn(const float* __restrict__ feat, const float* __restrict__ al,
                          const float* __restrict__ ar, float* __restrict__ el,
                          float* __restrict__ er, int N, int H, int D) {
  int idx = blockIdx.x * blockDim.x + threadIdx.x;
  if (idx >= N * H) return;
  int n = idx / H, h = idx - n * H;
  const float* f = feat + (size_t)n * H * D + (size_t)h * D;
  float sl = 0.f, sr = 0.f;
  for (int d = 0; d < D; ++d) { float x = f[d]; sl += x * al[h * D + d]; sr += x * ar[h * D + d]; }
  el[idx] = sl; er[idx] = sr;
}

// ---------------- segment-softmax: max then exp-sum (atomics into L2) ----------------
__global__ void edge_max(const int* __restrict__ src, const int* __restrict__ dst,
                         const float* __restrict__ el, const float* __restrict__ er,
                         float* __restrict__ m, long EH, int H) {
  long idx = (long)blockIdx.x * blockDim.x + threadIdx.x;
  if (idx >= EH) return;
  int e = (int)(idx / H), h = (int)(idx - (long)e * H);
  float ev = el[src[e] * H + h] + er[dst[e] * H + h];
  ev = ev > 0.f ? ev : NEG_SLOPE * ev;
  atomicMaxF(&m[dst[e] * H + h], ev);
}

__global__ void edge_expsum(const int* __restrict__ src, const int* __restrict__ dst,
                            const float* __restrict__ el, const float* __restrict__ er,
                            const float* __restrict__ m, float* __restrict__ s, long EH, int H) {
  long idx = (long)blockIdx.x * blockDim.x + threadIdx.x;
  if (idx >= EH) return;
  int e = (int)(idx / H), h = (int)(idx - (long)e * H);
  int dn = dst[e];
  float ev = el[src[e] * H + h] + er[dn * H + h];
  ev = ev > 0.f ? ev : NEG_SLOPE * ev;
  atomicAdd(&s[dn * H + h], __expf(ev - m[dn * H + h]));
}

// ---------------- message aggregation, layer1 (H=4, D=32): one wave per edge ----------------
__global__ __launch_bounds__(256)
void edge_agg_h4(const int* __restrict__ src, const int* __restrict__ dst,
                 const float* __restrict__ el, const float* __restrict__ er,
                 const float* __restrict__ m, const float* __restrict__ s,
                 const float* __restrict__ feat, float* __restrict__ agg, int E) {
  long gid = (long)blockIdx.x * blockDim.x + threadIdx.x;
  int e = (int)(gid >> 5);
  if (e >= E) return;
  int lane = (int)(gid & 31);
  int sn = src[e], dn = dst[e];
  if (e + 8 < E) __builtin_prefetch(feat + (size_t)src[e + 8] * 128, 0, 0);  // global_prefetch_b8
  int h = lane >> 3;                 // 4 heads x 8 lanes
  int c = h * 32 + (lane & 7) * 4;   // 4 consecutive channels per lane
  float ev = el[sn * 4 + h] + er[dn * 4 + h];
  ev = ev > 0.f ? ev : NEG_SLOPE * ev;
  float alpha = __expf(ev - m[dn * 4 + h]) / s[dn * 4 + h];
  const float4 f = *reinterpret_cast<const float4*>(feat + (size_t)sn * 128 + c);
  float* o = agg + (size_t)dn * 128 + c;
  atomicAdd(o + 0, f.x * alpha);
  atomicAdd(o + 1, f.y * alpha);
  atomicAdd(o + 2, f.z * alpha);
  atomicAdd(o + 3, f.w * alpha);
}

// ---------------- message aggregation, layer2 (H=1, D=16): 16 threads per edge ----------------
__global__ void edge_agg_h1(const int* __restrict__ src, const int* __restrict__ dst,
                            const float* __restrict__ el, const float* __restrict__ er,
                            const float* __restrict__ m, const float* __restrict__ s,
                            const float* __restrict__ feat, float* __restrict__ out, int E) {
  long gid = (long)blockIdx.x * blockDim.x + threadIdx.x;
  int e = (int)(gid >> 4);
  if (e >= E) return;
  int d = (int)(gid & 15);
  int sn = src[e], dn = dst[e];
  float ev = el[sn] + er[dn];
  ev = ev > 0.f ? ev : NEG_SLOPE * ev;
  float alpha = __expf(ev - m[dn]) / s[dn];
  atomicAdd(out + (size_t)dn * 16 + d, feat[(size_t)sn * 16 + d] * alpha);
}

// ---------------- ELU + bf16 convert (layer-1 output -> layer-2 input) ----------------
__global__ void elu_cvt(const float* __restrict__ in, unsigned short* __restrict__ out, long n) {
  long i = (long)blockIdx.x * blockDim.x + threadIdx.x;
  if (i < n) {
    float v = in[i];
    v = v > 0.f ? v : (__expf(v) - 1.f);
    out[i] = f2bf(v);
  }
}

extern "C" void kernel_launch(void* const* d_in, const int* in_sizes, int n_in,
                              void* d_out, int out_size, void* d_ws, size_t ws_size,
                              hipStream_t stream) {
  const float* x    = (const float*)d_in[0];
  const int*   esrc = (const int*)d_in[1];
  const int*   edst = (const int*)d_in[2];
  const float* W1   = (const float*)d_in[3];
  const float* al1  = (const float*)d_in[4];
  const float* ar1  = (const float*)d_in[5];
  const float* W2   = (const float*)d_in[6];
  const float* al2  = (const float*)d_in[7];
  const float* ar2  = (const float*)d_in[8];
  float* out = (float*)d_out;
  (void)in_sizes; (void)n_in; (void)out_size; (void)ws_size;

  const int N = NNODES, E = NEDGES;
  char* ws = (char*)d_ws;
  size_t off = 0;
  auto take = [&](size_t bytes) -> char* {
    char* p = ws + off;
    off = (off + bytes + 255) & ~(size_t)255;
    return p;
  };
  unsigned short* xbf  = (unsigned short*)take((size_t)N * 128 * 2);
  unsigned short* w1t  = (unsigned short*)take((size_t)128 * 128 * 2);
  unsigned short* w2t  = (unsigned short*)take((size_t)16 * 128 * 2);
  float* feat1 = (float*)take((size_t)N * 128 * 4);
  float* el1   = (float*)take((size_t)N * 4 * 4);
  float* er1   = (float*)take((size_t)N * 4 * 4);
  float* m1    = (float*)take((size_t)N * 4 * 4);
  float* s1    = (float*)take((size_t)N * 4 * 4);
  float* agg1  = (float*)take((size_t)N * 128 * 4);
  unsigned short* h1bf = (unsigned short*)take((size_t)N * 128 * 2);
  float* feat2 = (float*)take((size_t)N * 16 * 4);
  float* el2   = (float*)take((size_t)N * 4);
  float* er2   = (float*)take((size_t)N * 4);
  float* m2    = (float*)take((size_t)N * 4);
  float* s2    = (float*)take((size_t)N * 4);

  const int TB = 256;
  auto nb = [](long n, int tb) { return (unsigned)((n + tb - 1) / tb); };
  const float NEG_INF = -__builtin_inff();

  // ---- prep: bf16 conversions ----
  cvt_bf16<<<nb((long)N * 128, TB), TB, 0, stream>>>(x, xbf, (long)N * 128);
  transpose_cvt<<<nb(128 * 128, TB), TB, 0, stream>>>(W1, w1t, 128, 128);
  transpose_cvt<<<nb(128 * 16, TB), TB, 0, stream>>>(W2, w2t, 128, 16);

  // ---- init softmax state + accumulators (ws/out are poisoned) ----
  fill_f32<<<nb((long)N * 4, TB), TB, 0, stream>>>(m1, NEG_INF, (long)N * 4);
  fill_f32<<<nb((long)N * 4, TB), TB, 0, stream>>>(s1, 0.f, (long)N * 4);
  fill_f32<<<nb((long)N * 128, TB), TB, 0, stream>>>(agg1, 0.f, (long)N * 128);
  fill_f32<<<nb((long)N, TB), TB, 0, stream>>>(m2, NEG_INF, (long)N);
  fill_f32<<<nb((long)N, TB), TB, 0, stream>>>(s2, 0.f, (long)N);
  fill_f32<<<nb((long)N * 16, TB), TB, 0, stream>>>(out, 0.f, (long)N * 16);

  // ---- layer 1 ----
  {
    dim3 grid((N / 16 + 3) / 4, 1);                 // 6250 row tiles, 4 waves/block
    gemm_bf16<8><<<grid, 128, 0, stream>>>(xbf, w1t, feat1, N, 128);
  }
  node_attn<<<nb((long)N * 4, TB), TB, 0, stream>>>(feat1, al1, ar1, el1, er1, N, 4, 32);
  edge_max<<<nb((long)E * 4, TB), TB, 0, stream>>>(esrc, edst, el1, er1, m1, (long)E * 4, 4);
  edge_expsum<<<nb((long)E * 4, TB), TB, 0, stream>>>(esrc, edst, el1, er1, m1, s1, (long)E * 4, 4);
  edge_agg_h4<<<nb((long)E * 32, TB), TB, 0, stream>>>(esrc, edst, el1, er1, m1, s1, feat1, agg1, E);
  elu_cvt<<<nb((long)N * 128, TB), TB, 0, stream>>>(agg1, h1bf, (long)N * 128);

  // ---- layer 2 ----
  {
    dim3 grid((N / 16 + 3) / 4, 1);
    gemm_bf16<1><<<grid, 128, 0, stream>>>(h1bf, w2t, feat2, N, 16);
  }
  node_attn<<<nb((long)N, TB), TB, 0, stream>>>(feat2, al2, ar2, el2, er2, N, 1, 16);
  edge_max<<<nb((long)E, TB), TB, 0, stream>>>(esrc, edst, el2, er2, m2, (long)E, 1);
  edge_expsum<<<nb((long)E, TB), TB, 0, stream>>>(esrc, edst, el2, er2, m2, s2, (long)E, 1);
  edge_agg_h1<<<nb((long)E * 16, TB), TB, 0, stream>>>(esrc, edst, el2, er2, m2, s2, feat2, out, E);
}
